// WanSelfAttention_28690381538061
// MI455X (gfx1250) — matrix operand order repaired and checked by reading, not compiled, not run on hardware
//
#include <hip/hip_runtime.h>
#include <hip/hip_bf16.h>

// WanSelfAttention for MI455X (gfx1250): bf16 WMMA pipeline, fp32 accumulate.
// DIM=2048, NH=16, HD=128, S=3072 (B=1). RoPE: C=64 pairs = CT(22)+CH(21)+CH(21).
// Attention stages K/V tiles into LDS with CDNA5 async-to-LDS loads (ASYNCcnt).

#define DIMC   2048
#define NHC    16
#define HDC    128
#define CPAIR  64
#define CTC    22
#define CHC    21
#define EPSV   1e-6f

typedef __attribute__((ext_vector_type(16))) __bf16 v16bf;
typedef __attribute__((ext_vector_type(8)))  __bf16 v8bf;
typedef __attribute__((ext_vector_type(8)))  float  v8f;

// CDNA5 async global->LDS copy: each active lane copies 16B from its global
// address to its own LDS byte offset. Tracked by ASYNCcnt.
__device__ __forceinline__ void async_load_b128(unsigned lds_off, const void* gaddr) {
    asm volatile("global_load_async_to_lds_b128 %0, %1, off"
                 :: "v"(lds_off), "v"(gaddr) : "memory");
}
__device__ __forceinline__ void wait_asynccnt0() {
    asm volatile("s_wait_asynccnt 0x0" ::: "memory");
}

// ---------------- conversion kernels ----------------

__global__ void cast_f32_bf16(const float* __restrict__ src, __bf16* __restrict__ dst, int n) {
    int i = blockIdx.x * blockDim.x + threadIdx.x;
    if (i < n) dst[i] = (__bf16)src[i];
}

// wt[n][k] = w[k][n]  (DIM x DIM), so B-fragments read contiguous K.
__global__ void transpose_f32_bf16(const float* __restrict__ w, __bf16* __restrict__ wt) {
    int idx = blockIdx.x * blockDim.x + threadIdx.x;
    int n = idx / DIMC, k = idx % DIMC;
    wt[idx] = (__bf16)w[(size_t)k * DIMC + n];
}

// ---------------- bf16 WMMA GEMM: C = A(MxK) * Bt^T(NxK) + bias ----------------
// Block: 8 waves, block tile 128(M) x 64(N); wave tile 16 x 64.

__global__ __launch_bounds__(256)
void gemm_bf16(const __bf16* __restrict__ A, const __bf16* __restrict__ Bt,
               const float* __restrict__ bias, float* __restrict__ Cout,
               int N, int K) {
    const int lane = threadIdx.x & 31;
    const int wave = threadIdx.x >> 5;
    const int lm   = lane & 15;
    const int hi   = lane >> 4;
    const int m0   = blockIdx.y * 128 + wave * 16;
    const int n0   = blockIdx.x * 64;

    const __bf16* arow = A + (size_t)(m0 + lm) * K;
    v8f acc[4] = {};

    for (int k0 = 0; k0 < K; k0 += 32) {
        __builtin_prefetch(arow + k0 + 128, 0, 0);   // global_prefetch_b8
        v16bf a;
        v8bf a0 = *(const v8bf*)(arow + k0 + 16 * hi);
        v8bf a1 = *(const v8bf*)(arow + k0 + 16 * hi + 8);
#pragma unroll
        for (int i = 0; i < 8; ++i) { a[i] = a0[i]; a[i + 8] = a1[i]; }
#pragma unroll
        for (int j = 0; j < 4; ++j) {
            const __bf16* brow = Bt + (size_t)(n0 + 16 * j + lm) * K + k0 + 16 * hi;
            v16bf b = *(const v16bf*)brow;
            acc[j] = __builtin_amdgcn_wmma_f32_16x16x32_bf16(
                false, a, false, b, (short)0, acc[j], false, false);
        }
    }
#pragma unroll
    for (int j = 0; j < 4; ++j) {
        int n = n0 + 16 * j + lm;
        float bv = bias[n];
#pragma unroll
        for (int r = 0; r < 8; ++r) {
            int m = m0 + r + 8 * hi;
            Cout[(size_t)m * N + n] = acc[j][r] + bv;
        }
    }
}

// ---------------- QKV epilogue: RMSNorm + 3D RoPE + layout/convert ----------------
// One block (256 thr) per token. Writes q,k -> [NH][S][HD] bf16 (q pre-scaled by
// 1/sqrt(HD)); v -> transposed [NH][HD][S] bf16 for contiguous PV B-fragments.

__global__ __launch_bounds__(256)
void qkv_epilogue(const float* __restrict__ qf, const float* __restrict__ kf,
                  const float* __restrict__ vf,
                  const float* __restrict__ nqw, const float* __restrict__ nkw,
                  const float* __restrict__ fcos, const float* __restrict__ fsin,
                  const int* __restrict__ pf, const int* __restrict__ ph,
                  const int* __restrict__ pw,
                  __bf16* __restrict__ qb, __bf16* __restrict__ kb,
                  __bf16* __restrict__ vtb, int S) {
    (void)pf;
    const int s   = blockIdx.x;
    const int tid = threadIdx.x;
    const float* q = qf + (size_t)s * DIMC;
    const float* k = kf + (size_t)s * DIMC;
    const float* v = vf + (size_t)s * DIMC;

    float sq = 0.f, sk = 0.f;
    for (int i = tid; i < DIMC; i += 256) {
        float a = q[i]; sq += a * a;
        float b = k[i]; sk += b * b;
    }
    __shared__ float redq[8], redk[8];
#pragma unroll
    for (int off = 16; off > 0; off >>= 1) {
        sq += __shfl_xor(sq, off, 32);
        sk += __shfl_xor(sk, off, 32);
    }
    if ((tid & 31) == 0) { redq[tid >> 5] = sq; redk[tid >> 5] = sk; }
    __syncthreads();
    float tq = 0.f, tk = 0.f;
#pragma unroll
    for (int i = 0; i < 8; ++i) { tq += redq[i]; tk += redk[i]; }
    const float invq = rsqrtf(tq * (1.0f / DIMC) + EPSV);
    const float invk = rsqrtf(tk * (1.0f / DIMC) + EPSV);

    const int Hh = ph[0], Ww = pw[0];
    const int fi  = s / (Hh * Ww);
    const int rem = s % (Hh * Ww);
    const int hg  = rem / Ww;
    const int wg  = rem % Ww;
    const float qscale = 0.08838834764831845f;  // 1/sqrt(128)

    for (int p = tid; p < DIMC / 2; p += 256) {
        int d0 = 2 * p, d1 = d0 + 1;
        int head = d0 / HDC;
        int c = (d0 % HDC) >> 1;
        int row = (c < CTC) ? fi : ((c < CTC + CHC) ? hg : wg);
        float co = fcos[row * CPAIR + c];
        float si = fsin[row * CPAIR + c];

        size_t base = ((size_t)head * S + s) * HDC + (d0 % HDC);
        // q: norm + rope + scale
        float re = q[d0] * invq * nqw[d0];
        float im = q[d1] * invq * nqw[d1];
        qb[base]     = (__bf16)((re * co - im * si) * qscale);
        qb[base + 1] = (__bf16)((re * si + im * co) * qscale);
        // k: norm + rope
        re = k[d0] * invk * nkw[d0];
        im = k[d1] * invk * nkw[d1];
        kb[base]     = (__bf16)(re * co - im * si);
        kb[base + 1] = (__bf16)(re * si + im * co);
        // v: transpose store [head][d][s]
        size_t vbase = ((size_t)head * HDC + (d0 % HDC)) * S + s;
        vtb[vbase]     = (__bf16)v[d0];
        vtb[vbase + S] = (__bf16)v[d1];
    }
}

// ---------------- flash attention ----------------
// Block: 4 waves, one head, 64 queries (16 per wave). K (32x128) and V^T (128x32)
// tiles staged once per block into LDS via async-to-LDS loads and shared by all
// 4 waves; per-wave P bounce buffer re-layouts softmax C-frags into A-frags.

__global__ __launch_bounds__(128)
void attention(const __bf16* __restrict__ qb, const __bf16* __restrict__ kb,
               const __bf16* __restrict__ vtb, const int* __restrict__ seq_lens,
               __bf16* __restrict__ ob, int S) {
    const int tid  = threadIdx.x;
    const int wv4  = tid >> 5;          // wave id 0..3
    const int lane = tid & 31;
    const int lm   = lane & 15;
    const int hi   = lane >> 4;
    const int head = blockIdx.y;
    const int q0   = blockIdx.x * 64 + wv4 * 16;
    const int kvlen = seq_lens[0];

    __shared__ __bf16 Kt[32 * HDC];      // 8 KB: [key][hd]
    __shared__ __bf16 Vt[HDC * 32];      // 8 KB: [hd][key]
    __shared__ __bf16 Pt[4][16 * 32];    // 4 KB: per-wave P tile

    const unsigned ktBase = (unsigned)(unsigned long long)(uintptr_t)&Kt[0];
    const unsigned vtBase = (unsigned)(unsigned long long)(uintptr_t)&Vt[0];

    // Q A-fragments for the 4 K-steps covering HD=128
    const __bf16* qrow = qb + ((size_t)head * S + q0 + lm) * HDC;
    v16bf aq[4];
#pragma unroll
    for (int t = 0; t < 4; ++t) {
        v8bf a0 = *(const v8bf*)(qrow + 32 * t + 16 * hi);
        v8bf a1 = *(const v8bf*)(qrow + 32 * t + 16 * hi + 8);
#pragma unroll
        for (int i = 0; i < 8; ++i) { aq[t][i] = a0[i]; aq[t][i + 8] = a1[i]; }
    }

    float mrow[8], lrow[8];
    v8f acc[8] = {};
#pragma unroll
    for (int r = 0; r < 8; ++r) { mrow[r] = -3.0e38f; lrow[r] = 0.f; }

    const __bf16* kh = kb + (size_t)head * S * HDC;
    const __bf16* vh = vtb + (size_t)head * HDC * S;

    for (int kb0 = 0; kb0 < S; kb0 += 32) {
        // ---- async stage K tile (512 x 16B) and V^T tile (512 x 16B) ----
#pragma unroll
        for (int i = 0; i < 4; ++i) {
            int c = tid + 128 * i;
            int key = c >> 4, koff = (c & 15) * 8;          // Kt: 16 chunks/row
            async_load_b128(ktBase + c * 16, kh + (size_t)(kb0 + key) * HDC + koff);
            int hd = c >> 2, so = (c & 3) * 8;              // Vt: 4 chunks/row
            async_load_b128(vtBase + c * 16, vh + (size_t)hd * S + kb0 + so);
        }
        wait_asynccnt0();
        __syncthreads();

        // ---- scores: two 16-key sub-tiles from LDS ----
        v8f sfr[2];
#pragma unroll
        for (int sub = 0; sub < 2; ++sub) {
            v8f sacc = {};
            const __bf16* krow = &Kt[(16 * sub + lm) * HDC + 16 * hi];
#pragma unroll
            for (int t = 0; t < 4; ++t) {
                v16bf b = *(const v16bf*)(krow + 32 * t);
                sacc = __builtin_amdgcn_wmma_f32_16x16x32_bf16(
                    false, aq[t], false, b, (short)0, sacc, false, false);
            }
            sfr[sub] = sacc;
        }

        // ---- online softmax ----
        const bool v0 = (kb0 + lm) < kvlen;
        const bool v1 = (kb0 + 16 + lm) < kvlen;
#pragma unroll
        for (int r = 0; r < 8; ++r) {
            float s0 = v0 ? sfr[0][r] : -3.0e38f;
            float s1 = v1 ? sfr[1][r] : -3.0e38f;
            float mx = fmaxf(s0, s1);
#pragma unroll
            for (int off = 1; off < 16; off <<= 1) mx = fmaxf(mx, __shfl_xor(mx, off, 32));
            float mnew = fmaxf(mrow[r], mx);
            float p0 = v0 ? __expf(s0 - mnew) : 0.f;
            float p1 = v1 ? __expf(s1 - mnew) : 0.f;
            float ladd = p0 + p1;
#pragma unroll
            for (int off = 1; off < 16; off <<= 1) ladd += __shfl_xor(ladd, off, 32);
            float sc = __expf(mrow[r] - mnew);
            mrow[r] = mnew;
            lrow[r] = lrow[r] * sc + ladd;
#pragma unroll
            for (int j = 0; j < 8; ++j) acc[j][r] *= sc;
            Pt[wv4][(r + 8 * hi) * 32 + lm]      = (__bf16)p0;
            Pt[wv4][(r + 8 * hi) * 32 + 16 + lm] = (__bf16)p1;
        }

        // ---- P as A-fragment (C->A re-layout through per-wave LDS tile) ----
        v16bf ap;
        {
            v8bf a0 = *(const v8bf*)(&Pt[wv4][lm * 32 + 16 * hi]);
            v8bf a1 = *(const v8bf*)(&Pt[wv4][lm * 32 + 16 * hi + 8])
#pragma unroll
            for (int i = 0; i < 8; ++i) { ap[i] = a0[i]; ap[i + 8] = a1[i]; }
        }
        // ---- PV accumulate from LDS V^T tile ----
#pragma unroll
        for (int j = 0; j < 8; ++j) {
            v16bf b = *(const v16bf*)&Vt[(16 * j + lm) * 32 + 16 * hi];
            acc[j] = __builtin_amdgcn_wmma_f32_16x16x32_bf16(
                false, ap, false, b, (short)0, acc[j], false, false);
        }
        __syncthreads();   // protect K/V tiles before next stage
    }

    // normalize and emit O[s][head*HD + d] bf16
#pragma unroll
    for (int r = 0; r < 8; ++r) {
        float inv = 1.f / lrow[r];
        int m = q0 + r + 8 * hi;
#pragma unroll
        for (int j = 0; j < 8; ++j) {
            ob[(size_t)m * DIMC + head * HDC + 16 * j + lm] = (__bf16)(acc[j][r] * inv);
        }
    }
}

// ---------------- host launcher ----------------

extern "C" void kernel_launch(void* const* d_in, const int* in_sizes, int n_in,
                              void* d_out, int out_size, void* d_ws, size_t ws_size,
                              hipStream_t stream) {
    (void)n_in; (void)out_size; (void)ws_size;
    const float* x    = (const float*)d_in[0];
    const float* wq   = (const float*)d_in[1];
    const float* bq   = (const float*)d_in[2];
    const float* wk   = (const float*)d_in[3];
    const float* bk   = (const float*)d_in[4];
    const float* wv   = (const float*)d_in[5];
    const float* bv   = (const float*)d_in[6];
    const float* wo   = (const float*)d_in[7];
    const float* bo   = (const float*)d_in[8];
    const float* nqw  = (const float*)d_in[9];
    const float* nkw  = (const float*)d_in[10];
    const float* fcos = (const float*)d_in[11];
    const float* fsin = (const float*)d_in[12];
    const int*   seqp = (const int*)d_in[13];
    const int*   pf   = (const int*)d_in[14];
    const int*   ph   = (const int*)d_in[15];
    const int*   pw   = (const int*)d_in[16];

    const int S = in_sizes[0] / DIMC;   // 3072

    char*  ws  = (char*)d_ws;
    size_t off = 0;
    auto alloc = [&](size_t bytes) -> char* {
        char* p = ws + off;
        off += (bytes + 255) & ~(size_t)255;
        return p;
    };
    __bf16* xb  = (__bf16*)alloc((size_t)S * DIMC * 2);
    __bf16* wqT = (__bf16*)alloc((size_t)DIMC * DIMC * 2);
    __bf16* wkT = (__bf16*)alloc((size_t)DIMC * DIMC * 2);
    __bf16* wvT = (__bf16*)alloc((size_t)DIMC * DIMC * 2);
    __bf16* woT = (__bf16*)alloc((size_t)DIMC * DIMC * 2);
    float*  qf  = (float*)alloc((size_t)S * DIMC * 4);
    float*  kf  = (float*)alloc((size_t)S * DIMC * 4);
    float*  vf  = (float*)alloc((size_t)S * DIMC * 4);
    __bf16* qbv = (__bf16*)alloc((size_t)S * DIMC * 2);
    __bf16* kbv = (__bf16*)alloc((size_t)S * DIMC * 2);
    __bf16* vtb = (__bf16*)alloc((size_t)S * DIMC * 2);
    __bf16* obf = (__bf16*)alloc((size_t)S * DIMC * 2);

    const int nx = S * DIMC;
    cast_f32_bf16<<<(nx + 255) / 256, 256, 0, stream>>>(x, xb, nx);
    const int nw = DIMC * DIMC;
    transpose_f32_bf16<<<nw / 256, 256, 0, stream>>>(wq, wqT);
    transpose_f32_bf16<<<nw / 256, 256, 0, stream>>>(wk, wkT);
    transpose_f32_bf16<<<nw / 256, 256, 0, stream>>>(wv, wvT);
    transpose_f32_bf16<<<nw / 256, 256, 0, stream>>>(wo, woT);

    dim3 ggrid(DIMC / 64, S / 128);
    gemm_bf16<<<ggrid, 256, 0, stream>>>(xb, wqT, bq, qf, DIMC, DIMC);
    gemm_bf16<<<ggrid, 256, 0, stream>>>(xb, wkT, bk, kf, DIMC, DIMC);
    gemm_bf16<<<ggrid, 256, 0, stream>>>(xb, wvT, bv, vf, DIMC, DIMC);

    qkv_epilogue<<<S, 256, 0, stream>>>(qf, kf, vf, nqw, nkw, fcos, fsin,
                                        pf, ph, pw, qbv, kbv, vtb, S);

    attention<<<dim3(S / 64, NHC), 128, 0, stream>>>(qbv, kbv, vtb, seqp, obf, S);

    gemm_bf16<<<ggrid, 256, 0, stream>>>(obf, woT, bo, (float*)d_out, DIMC, DIMC);
}
